// DRAGConv_17892833755548
// MI455X (gfx1250) — matrix operand order repaired and checked by reading, not compiled, not run on hardware
//
#include <hip/hip_runtime.h>
#include <hip/hip_bf16.h>
#include <math.h>

// Problem constants (match reference)
#define N_NODES 50000
#define N_EDGES 800000
#define IN_F    256
#define HEADS   4
#define HD      256      // HEADS * 64
#define DHEAD   64
#define NEG_SLOPE 0.2f

typedef __attribute__((ext_vector_type(16))) __bf16 v16bf;
typedef __attribute__((ext_vector_type(8)))  __bf16 v8bf;
typedef __attribute__((ext_vector_type(4)))  __bf16 v4bf;
typedef __attribute__((ext_vector_type(8)))  float  v8f;
typedef __attribute__((ext_vector_type(4)))  float  v4f;

// ---------------------------------------------------------------------------
// fp32 -> bf16 conversion, 4 elements/thread (float4 in, 8B packed out)
// ---------------------------------------------------------------------------
__global__ void f32_to_bf16_kernel(const float* __restrict__ src,
                                   __bf16* __restrict__ dst, long n4) {
    long i = (long)blockIdx.x * blockDim.x + threadIdx.x;
    if (i >= n4) return;
    v4f x = *(const v4f*)(src + i * 4);
    v4bf y;
    y.x = (__bf16)x.x; y.y = (__bf16)x.y; y.z = (__bf16)x.z; y.w = (__bf16)x.w;
    *(v4bf*)(dst + i * 4) = y;
}

// ---------------------------------------------------------------------------
// init: segmax = -inf, denom = 0, out = 0
// ---------------------------------------------------------------------------
__global__ void init_kernel(float* __restrict__ segmax, float* __restrict__ denom,
                            float* __restrict__ out, int n4, long n256) {
    long i = (long)blockIdx.x * blockDim.x + threadIdx.x;
    if (i < n4) { segmax[i] = -INFINITY; denom[i] = 0.0f; }
    if (i < n256) out[i] = 0.0f;
}

// ---------------------------------------------------------------------------
// Projection GEMM: Y[N,256] = X[N,256] @ W[256,256]^T + bias
// bf16 inputs, f32 accumulate via v_wmma_f32_16x16x32_bf16.
// One wave -> 16(M) x 64(N) tile; 8 waves/block -> 128 rows x 64 cols.
// Software-pipelined: next K-step's A/B loads are issued before the current
// K-step's WMMAs so VMEM overlaps the matrix pipe (no full loadcnt drains).
//
// A operand (contiguous k order, same as B): lane(l) row = base + l%16;
//   elems 0..7  = K[kb + 16*(l/16)     .. +7]  (contiguous)
//   elems 8..15 = K[kb + 16*(l/16) + 8 .. +7]  (contiguous)
// B operand: lane(l) col = colBase + l%16;
//   elems 0..15 = K[kb + 16*(l/16) .. +15]     (contiguous v16bf)
//   which is exactly row-major W[col][k..k+15].
// ---------------------------------------------------------------------------
__global__ __launch_bounds__(256) void proj_gemm_kernel(
    const __bf16* __restrict__ Xb,    // [N, 256] bf16
    const __bf16* __restrict__ Wb,    // [256, 256] bf16, row-major [out, k]
    const float*  __restrict__ bias,  // [256]
    float* __restrict__ Y,            // [N, 256] f32
    int nrows)
{
    const int wave = threadIdx.x >> 5;
    const int lane = threadIdx.x & 31;
    const int half = lane >> 4;
    const int m    = lane & 15;

    const int rowBase = blockIdx.x * 128 + wave * 16;
    if (rowBase >= nrows) return;                 // wave-uniform: EXEC stays all-1s
    const int colBase = blockIdx.y * 64;

    const __bf16* aRow  = Xb + (size_t)(rowBase + m) * IN_F + 16 * half;
    const __bf16* bCol0 = Wb + (size_t)(colBase +  0 + m) * IN_F + 16 * half;
    const __bf16* bCol1 = Wb + (size_t)(colBase + 16 + m) * IN_F + 16 * half;
    const __bf16* bCol2 = Wb + (size_t)(colBase + 32 + m) * IN_F + 16 * half;
    const __bf16* bCol3 = Wb + (size_t)(colBase + 48 + m) * IN_F + 16 * half;

    v8f acc0 = {}, acc1 = {}, acc2 = {}, acc3 = {};

    // ---- prologue: stage K-step 0 -----------------------------------------
    v8bf  a0 = *(const v8bf*)(aRow);
    v8bf  a1 = *(const v8bf*)(aRow + 8);
    v16bf nB0 = *(const v16bf*)(bCol0);
    v16bf nB1 = *(const v16bf*)(bCol1);
    v16bf nB2 = *(const v16bf*)(bCol2);
    v16bf nB3 = *(const v16bf*)(bCol3);

    #pragma unroll
    for (int kb = 0; kb < IN_F; kb += 32) {
        union { v16bf v; v8bf h[2]; } A;
        A.h[0] = a0;  A.h[1] = a1;
        v16bf B0 = nB0, B1 = nB1, B2 = nB2, B3 = nB3;

        if (kb + 32 < IN_F) {            // issue next-step loads first
            a0  = *(const v8bf*) (aRow  + kb + 32);
            a1  = *(const v8bf*) (aRow  + kb + 40);
            nB0 = *(const v16bf*)(bCol0 + kb + 32);
            nB1 = *(const v16bf*)(bCol1 + kb + 32);
            nB2 = *(const v16bf*)(bCol2 + kb + 32);
            nB3 = *(const v16bf*)(bCol3 + kb + 32);
        }

        acc0 = __builtin_amdgcn_wmma_f32_16x16x32_bf16(false, A.v, false, B0, (short)0, acc0, false, false);
        acc1 = __builtin_amdgcn_wmma_f32_16x16x32_bf16(false, A.v, false, B1, (short)0, acc1, false, false);
        acc2 = __builtin_amdgcn_wmma_f32_16x16x32_bf16(false, A.v, false, B2, (short)0, acc2, false, false);
        acc3 = __builtin_amdgcn_wmma_f32_16x16x32_bf16(false, A.v, false, B3, (short)0, acc3, false, false);
    }

    // C/D layout: VGPR r, lane l -> (M = r + 8*(l/16), N = l%16)
    #define STORE_TILE(ACC, J)                                                  \
    {                                                                           \
        const int col = colBase + 16 * (J) + m;                                 \
        const float bj = bias[col];                                             \
        float* yp = Y + (size_t)(rowBase + 8 * half) * HD + col;                \
        _Pragma("unroll")                                                       \
        for (int r = 0; r < 8; ++r) yp[(size_t)r * HD] = (ACC)[r] + bj;         \
    }
    STORE_TILE(acc0, 0)
    STORE_TILE(acc1, 1)
    STORE_TILE(acc2, 2)
    STORE_TILE(acc3, 3)
    #undef STORE_TILE
}

// ---------------------------------------------------------------------------
// Sign-aware float atomic max via integer atomics (works for any sign mix)
// ---------------------------------------------------------------------------
__device__ __forceinline__ void atomicMaxF32(float* addr, float val) {
    if (val >= 0.0f) atomicMax((int*)addr, __float_as_int(val));
    else             atomicMin((unsigned int*)addr, __float_as_uint(val));
}

// ---------------------------------------------------------------------------
// Edge logits: logits[e,h] = sum_d LR(xs[src,h,d] + xd[dst,h,d]) * attn[h,d]
// One wave per edge; lane covers d = lane and d = lane+32 per head.
// Also atomic-max into segmax[dst,h].
// ---------------------------------------------------------------------------
__global__ __launch_bounds__(256) void edge_logits_kernel(
    const float* __restrict__ xs, const float* __restrict__ xd,
    const float* __restrict__ attn,
    const int* __restrict__ src, const int* __restrict__ dst,
    float* __restrict__ logits, float* __restrict__ segmax, int nE)
{
    const int e = blockIdx.x * 8 + (threadIdx.x >> 5);
    if (e >= nE) return;
    const int lane = threadIdx.x & 31;
    const int u = src[e], v = dst[e];
    const float* ps = xs + (size_t)u * HD;
    const float* pd = xd + (size_t)v * HD;

    #pragma unroll
    for (int h = 0; h < HEADS; ++h) {
        const int i0 = h * DHEAD + lane, i1 = i0 + 32;
        float t0 = ps[i0] + pd[i0];
        float t1 = ps[i1] + pd[i1];
        t0 = (t0 > 0.0f) ? t0 : NEG_SLOPE * t0;
        t1 = (t1 > 0.0f) ? t1 : NEG_SLOPE * t1;
        float s = t0 * attn[i0] + t1 * attn[i1];
        #pragma unroll
        for (int off = 16; off > 0; off >>= 1) s += __shfl_xor(s, off, 32);
        if (lane == 0) {
            logits[(size_t)e * HEADS + h] = s;
            atomicMaxF32(&segmax[(size_t)v * HEADS + h], s);
        }
    }
}

// ---------------------------------------------------------------------------
// exp(logit - segmax[dst]) in place, accumulate denom[dst,h]
// ---------------------------------------------------------------------------
__global__ void edge_exp_kernel(const float* __restrict__ segmax,
                                const int* __restrict__ dst,
                                float* __restrict__ logits,
                                float* __restrict__ denom, int nEH)
{
    const int i = blockIdx.x * blockDim.x + threadIdx.x;
    if (i >= nEH) return;
    const int e = i >> 2, h = i & 3;
    const int v = dst[e];
    const float ex = __expf(logits[i] - segmax[v * HEADS + h]);
    logits[i] = ex;
    atomicAdd(&denom[v * HEADS + h], ex);
}

// ---------------------------------------------------------------------------
// Aggregation: out[dst,h,d] += (ex[e,h]/denom[dst,h]) * xv[src,h,d]
// One wave per edge; 8 f32 atomics per lane.
// ---------------------------------------------------------------------------
__global__ __launch_bounds__(256) void aggregate_kernel(
    const float* __restrict__ xv, const float* __restrict__ ex,
    const float* __restrict__ denom,
    const int* __restrict__ src, const int* __restrict__ dst,
    float* __restrict__ out, int nE)
{
    const int e = blockIdx.x * 8 + (threadIdx.x >> 5);
    if (e >= nE) return;
    const int lane = threadIdx.x & 31;
    const int u = src[e], v = dst[e];
    const float* pv = xv + (size_t)u * HD;
    float* po = out + (size_t)v * HD;

    #pragma unroll
    for (int h = 0; h < HEADS; ++h) {
        const float a = ex[(size_t)e * HEADS + h] / denom[(size_t)v * HEADS + h];
        const int i0 = h * DHEAD + lane, i1 = i0 + 32;
        atomicAdd(&po[i0], a * pv[i0]);
        atomicAdd(&po[i1], a * pv[i1]);
    }
}

// ---------------------------------------------------------------------------
// Host-side orchestration
// ---------------------------------------------------------------------------
extern "C" void kernel_launch(void* const* d_in, const int* in_sizes, int n_in,
                              void* d_out, int out_size, void* d_ws, size_t ws_size,
                              hipStream_t stream)
{
    const float* feat = (const float*)d_in[0];
    const float* Wsrc = (const float*)d_in[1];
    const float* bsrc = (const float*)d_in[2];
    const float* Wdst = (const float*)d_in[3];
    const float* bdst = (const float*)d_in[4];
    const float* Wv   = (const float*)d_in[5];
    const float* bv   = (const float*)d_in[6];
    const float* attn = (const float*)d_in[7];
    const int*   src  = (const int*)d_in[8];
    const int*   dst  = (const int*)d_in[9];
    float* out = (float*)d_out;

    // ---- workspace layout -------------------------------------------------
    char* w = (char*)d_ws;
    const size_t nodeF = (size_t)N_NODES * HD;          // 12.8M floats
    float*  xs     = (float*)w;                w += nodeF * sizeof(float);
    float*  xd     = (float*)w;                w += nodeF * sizeof(float);
    float*  xv     = (float*)w;                w += nodeF * sizeof(float);
    float*  logits = (float*)w;                w += (size_t)N_EDGES * HEADS * sizeof(float);
    float*  segmax = (float*)w;                w += (size_t)N_NODES * HEADS * sizeof(float);
    float*  denom  = (float*)w;                w += (size_t)N_NODES * HEADS * sizeof(float);
    __bf16* featb  = (__bf16*)w;               w += nodeF * sizeof(__bf16);
    __bf16* Wb0    = (__bf16*)w;               w += (size_t)HD * IN_F * sizeof(__bf16);
    __bf16* Wb1    = (__bf16*)w;               w += (size_t)HD * IN_F * sizeof(__bf16);
    __bf16* Wb2    = (__bf16*)w;               /* end */

    const long nFeat4 = (long)N_NODES * IN_F / 4;
    const long nW4    = (long)HD * IN_F / 4;

    // ---- precision conversion (4 elems/thread) ----------------------------
    f32_to_bf16_kernel<<<(nFeat4 + 255) / 256, 256, 0, stream>>>(feat, featb, nFeat4);
    f32_to_bf16_kernel<<<(nW4 + 255) / 256, 256, 0, stream>>>(Wsrc, Wb0, nW4);
    f32_to_bf16_kernel<<<(nW4 + 255) / 256, 256, 0, stream>>>(Wdst, Wb1, nW4);
    f32_to_bf16_kernel<<<(nW4 + 255) / 256, 256, 0, stream>>>(Wv,   Wb2, nW4);

    // ---- init segmax/denom/out --------------------------------------------
    init_kernel<<<(nodeF + 255) / 256, 256, 0, stream>>>(
        segmax, denom, out, N_NODES * HEADS, (long)nodeF);

    // ---- three WMMA projections -------------------------------------------
    dim3 gGemm((N_NODES + 127) / 128, HD / 64);
    proj_gemm_kernel<<<gGemm, 256, 0, stream>>>(featb, Wb0, bsrc, xs, N_NODES);
    proj_gemm_kernel<<<gGemm, 256, 0, stream>>>(featb, Wb1, bdst, xd, N_NODES);
    proj_gemm_kernel<<<gGemm, 256, 0, stream>>>(featb, Wb2, bv,   xv, N_NODES);

    // ---- edge phase --------------------------------------------------------
    edge_logits_kernel<<<(N_EDGES + 7) / 8, 256, 0, stream>>>(
        xs, xd, attn, src, dst, logits, segmax, N_EDGES);
    edge_exp_kernel<<<(N_EDGES * HEADS + 255) / 256, 256, 0, stream>>>(
        segmax, dst, logits, denom, N_EDGES * HEADS);
    aggregate_kernel<<<(N_EDGES + 7) / 8, 256, 0, stream>>>(
        xv, logits, denom, src, dst, out, N_EDGES);
}
